// WanAttention_65721589563939
// MI455X (gfx1250) — compile-verified, matrix operand order checked by reading, not run on hardware
//
#include <hip/hip_runtime.h>
#include <math.h>

typedef __attribute__((ext_vector_type(16))) _Float16 v16h;
typedef __attribute__((ext_vector_type(8)))  float    v8f;

#define S_    4096
#define DIM_  2048
#define H_    16
#define DH_   128

// ---------------------------------------------------------------------------
// WMMA fragment loaders, per CDNA5 ISA 7.12.2 (wave32).
// A (16x32 f16): lane L holds row m=L%16; halves map to K via
//   VGPR v, pos p -> K = (v>=4?16:0) + (v%4)*2 + p + 8*(L/16)
// B (32x16 f16): lane L holds col n=L%16; same K mapping, strided over rows.
// ---------------------------------------------------------------------------
__device__ __forceinline__ v16h lds_load_fragA(const _Float16* base, int ld) {
  const int lane = threadIdx.x & 31;
  const int m    = lane & 15;
  const int hi8  = (lane >> 4) << 3;
  v16h f;
#pragma unroll
  for (int v = 0; v < 8; ++v) {
    const int k0 = ((v & 4) << 2) + ((v & 3) << 1) + hi8;
    f[2 * v]     = base[m * ld + k0];
    f[2 * v + 1] = base[m * ld + k0 + 1];
  }
  return f;
}

__device__ __forceinline__ v16h lds_load_fragB(const _Float16* base, int ld) {
  const int lane = threadIdx.x & 31;
  const int n    = lane & 15;
  const int hi8  = (lane >> 4) << 3;
  v16h f;
#pragma unroll
  for (int v = 0; v < 8; ++v) {
    const int k0 = ((v & 4) << 2) + ((v & 3) << 1) + hi8;
    f[2 * v]     = base[k0 * ld + n];
    f[2 * v + 1] = base[(k0 + 1) * ld + n];
  }
  return f;
}

__device__ __forceinline__ v8f wmma_f16(v16h a, v16h b, v8f c) {
  return __builtin_amdgcn_wmma_f32_16x16x32_f16(false, a, false, b,
                                                (short)0, c, false, false);
}

// ---------------------------------------------------------------------------
// GEMM: C[M,N] = A[M,K] * B[K,N] + bias[N].
// 256 threads = 8 waves; block tile 128x128; wave tile 64x32; K-step 64
// (16 v_wmma per barrier pair).  All global staging is 16-byte (b128).
// ---------------------------------------------------------------------------
template <bool A_F16>
__global__ __launch_bounds__(256) void gemm_bias_kernel(
    const void* __restrict__ Ain, const float* __restrict__ Bw,
    const float* __restrict__ bias, float* __restrict__ Cout,
    int M, int N, int K)
{
  __shared__ _Float16 As[128 * 72];   // 128 rows x 64 K (+8 pad; 144B pitch)
  __shared__ _Float16 Bs[64 * 136];   // 64 K x 128 cols (+8 pad)

  const int tid  = threadIdx.x;
  const int wave = tid >> 5;
  const int lane = tid & 31;
  const int wm   = (wave & 1) * 64;
  const int wn   = (wave >> 1) * 32;
  const int m0   = blockIdx.y * 128;
  const int n0   = blockIdx.x * 128;

  const float*    Af = (const float*)Ain;
  const _Float16* Ah = (const _Float16*)Ain;

  v8f acc[4][2] = {};

  for (int k0 = 0; k0 < K; k0 += 64) {
    __syncthreads();
    // ---- stage A tile (128 x 64) ----
    if (A_F16) {
      // copy 16B chunks (8 halves); 1024 chunks / 256 threads = 4 each
#pragma unroll
      for (int pass = 0; pass < 4; ++pass) {
        const int chunk = pass * 256 + tid;
        const int r  = chunk >> 3;
        const int cc = (chunk & 7) * 8;
        *(float4*)(As + r * 72 + cc) =
            *(const float4*)(Ah + (size_t)(m0 + r) * K + k0 + cc);
      }
    } else {
      // float4 loads + f32->f16 convert; 8 passes of 256x4 floats
#pragma unroll
      for (int pass = 0; pass < 8; ++pass) {
        const int r = pass * 16 + (tid >> 4);
        const int c = (tid & 15) * 4;
        const float4 v =
            *(const float4*)(Af + (size_t)(m0 + r) * K + k0 + c);
        As[r * 72 + c + 0] = (_Float16)v.x;
        As[r * 72 + c + 1] = (_Float16)v.y;
        As[r * 72 + c + 2] = (_Float16)v.z;
        As[r * 72 + c + 3] = (_Float16)v.w;
      }
    }
    // ---- stage B tile (64 x 128), f32 weights -> f16 ----
#pragma unroll
    for (int pass = 0; pass < 8; ++pass) {
      const int kr = pass * 8 + (tid >> 5);
      const int c  = (tid & 31) * 4;
      const float4 v =
          *(const float4*)(Bw + (size_t)(k0 + kr) * N + n0 + c);
      Bs[kr * 136 + c + 0] = (_Float16)v.x;
      Bs[kr * 136 + c + 1] = (_Float16)v.y;
      Bs[kr * 136 + c + 2] = (_Float16)v.z;
      Bs[kr * 136 + c + 3] = (_Float16)v.w;
    }
    if (k0 + 64 < K) {
      if (A_F16)
        __builtin_prefetch(Ah + (size_t)(m0 + (tid >> 3)) * K + k0 + 64, 0, 1);
      else
        __builtin_prefetch(Af + (size_t)(m0 + (tid >> 4)) * K + k0 + 64, 0, 1);
      __builtin_prefetch(Bw + (size_t)(k0 + 64 + (tid >> 5)) * N + n0, 0, 1);
    }
    __syncthreads();

    // ---- 16 WMMAs: 4 m-tiles x 2 n-tiles x 2 K-halves ----
    v16h bfr[2][2];
#pragma unroll
    for (int kf = 0; kf < 2; ++kf)
#pragma unroll
      for (int nt = 0; nt < 2; ++nt)
        bfr[kf][nt] = lds_load_fragB(Bs + kf * 32 * 136 + (wn + nt * 16), 136);
#pragma unroll
    for (int mt = 0; mt < 4; ++mt) {
#pragma unroll
      for (int kf = 0; kf < 2; ++kf) {
        v16h afr = lds_load_fragA(As + (wm + mt * 16) * 72 + kf * 32, 72);
#pragma unroll
        for (int nt = 0; nt < 2; ++nt)
          acc[mt][nt] = wmma_f16(afr, bfr[kf][nt], acc[mt][nt]);
      }
    }
  }

  // epilogue: C layout -> global (VGPR r: row = r + 8*(lane/16); col = lane%16)
  const int hi8 = (lane >> 4) << 3;
  const int nc  = lane & 15;
#pragma unroll
  for (int mt = 0; mt < 4; ++mt)
#pragma unroll
    for (int nt = 0; nt < 2; ++nt)
#pragma unroll
      for (int r = 0; r < 8; ++r) {
        const int row = m0 + wm + mt * 16 + r + hi8;
        const int col = n0 + wn + nt * 16 + nc;
        Cout[(size_t)row * N + col] = acc[mt][nt][r] + bias[col];
      }
}

// ---------------------------------------------------------------------------
// RMSNorm over DIM + RoPE per head; writes f16 for the WMMA attention stage.
// One 256-thread block (8 wave32) per sequence row.
// ---------------------------------------------------------------------------
__global__ __launch_bounds__(256) void rmsnorm_rope_kernel(
    const float* __restrict__ X, const float* __restrict__ g,
    const float* __restrict__ fc, const float* __restrict__ fs,
    _Float16* __restrict__ Y)
{
  __shared__ float red[8];
  const int s = blockIdx.x;
  const float* row = X + (size_t)s * DIM_;

  float ss = 0.f;
  for (int i = threadIdx.x; i < DIM_; i += 256) { float v = row[i]; ss += v * v; }
#pragma unroll
  for (int off = 16; off > 0; off >>= 1) ss += __shfl_down(ss, off); // wave32
  if ((threadIdx.x & 31) == 0) red[threadIdx.x >> 5] = ss;
  __syncthreads();
  float tot = 0.f;
#pragma unroll
  for (int i = 0; i < 8; ++i) tot += red[i];
  const float rstd = rsqrtf(tot * (1.0f / DIM_) + 1e-5f);

  const float* fcs = fc + (size_t)s * DH_;
  const float* fss = fs + (size_t)s * DH_;
  for (int p = threadIdx.x; p < DIM_ / 2; p += 256) {
    const int head = p >> 6, i = p & 63;
    const int d0 = head * DH_ + 2 * i;
    const float x1 = row[d0]     * rstd * g[d0];
    const float x2 = row[d0 + 1] * rstd * g[d0 + 1];
    const float c  = fcs[2 * i];        // cos = freqs_cos[..., 0::2]
    const float sn = fss[2 * i + 1];    // sin = freqs_sin[..., 1::2]
    Y[(size_t)s * DIM_ + d0]     = (_Float16)(x1 * c - x2 * sn);
    Y[(size_t)s * DIM_ + d0 + 1] = (_Float16)(x1 * sn + x2 * c);
  }
}

__global__ void f32_to_f16_kernel(const float* __restrict__ in,
                                  _Float16* __restrict__ out, int n) {
  for (int i = blockIdx.x * blockDim.x + threadIdx.x; i < n;
       i += gridDim.x * blockDim.x)
    out[i] = (_Float16)in[i];
}

// ---------------------------------------------------------------------------
// Flash attention: grid (S/64, H); 128 threads = 4 waves; each wave owns 16
// queries.  Key blocks of 32 stream through LDS.  QK^T uses the A-loader on
// the K tile (row-major [key][dh] == transposed-B fragment).  Online softmax
// via 16-lane xor-shuffle row reductions (wave32).  P goes through LDS to
// reach A-fragment layout for the P*V WMMAs.
// ---------------------------------------------------------------------------
__global__ __launch_bounds__(128) void flash_attn_kernel(
    const _Float16* __restrict__ Qh, const _Float16* __restrict__ Kh,
    const _Float16* __restrict__ Vh, _Float16* __restrict__ Oh)
{
  __shared__ _Float16 Qs[64 * 136];
  __shared__ _Float16 Ks[32 * 136];
  __shared__ _Float16 Vs[32 * 136];
  __shared__ _Float16 Ps[4 * 16 * 40];

  const int tid  = threadIdx.x;
  const int wave = tid >> 5;
  const int lane = tid & 31;
  const int hi8  = (lane >> 4) << 3;
  const int nc   = lane & 15;
  const int h    = blockIdx.y;
  const int s0   = blockIdx.x * 64;
  const int q0w  = wave * 16;
  const float scale = 0.08838834764831845f;  // 1/sqrt(128)

  for (int i = tid; i < 64 * 16; i += 128) {  // 16B chunks: 64 rows x 16 chunks
    const int r = i >> 4, cc = (i & 15) * 8;
    *(float4*)(Qs + r * 136 + cc) =
        *(const float4*)(Qh + (size_t)(s0 + r) * DIM_ + h * DH_ + cc);
  }

  v8f Oacc[8] = {};
  float mrow[8], lrow[8];
#pragma unroll
  for (int r = 0; r < 8; ++r) { mrow[r] = -1e30f; lrow[r] = 0.0f; }

  for (int kb = 0; kb < S_ / 32; ++kb) {
    __syncthreads();
    for (int i = tid; i < 32 * 16; i += 128) {  // 16B chunks
      const int r = i >> 4, cc = (i & 15) * 8;
      *(float4*)(Ks + r * 136 + cc) =
          *(const float4*)(Kh + (size_t)(kb * 32 + r) * DIM_ + h * DH_ + cc);
      *(float4*)(Vs + r * 136 + cc) =
          *(const float4*)(Vh + (size_t)(kb * 32 + r) * DIM_ + h * DH_ + cc);
    }
    if (kb + 1 < S_ / 32) {
      __builtin_prefetch(&Kh[(size_t)((kb + 1) * 32) * DIM_ + h * DH_], 0, 1);
      __builtin_prefetch(&Vh[(size_t)((kb + 1) * 32) * DIM_ + h * DH_], 0, 1);
    }
    __syncthreads();

    // S = Q * K^T : two 16-key subtiles, 4 WMMAs over dh each
    v8f sacc[2] = {};
#pragma unroll
    for (int nt = 0; nt < 2; ++nt)
#pragma unroll
      for (int dk = 0; dk < 4; ++dk) {
        v16h a = lds_load_fragA(Qs + q0w * 136 + dk * 32, 136);
        v16h b = lds_load_fragA(Ks + (nt * 16) * 136 + dk * 32, 136);
        sacc[nt] = wmma_f16(a, b, sacc[nt]);
      }

    // online softmax per owned row (r + 8*(lane/16)); row-reduce over 16 lanes
#pragma unroll
    for (int r = 0; r < 8; ++r) {
      const float s0v = sacc[0][r] * scale;
      const float s1v = sacc[1][r] * scale;
      float v = fmaxf(s0v, s1v);
#pragma unroll
      for (int off = 1; off < 16; off <<= 1) v = fmaxf(v, __shfl_xor(v, off));
      const float mnew = fmaxf(mrow[r], v);
      const float corr = __expf(mrow[r] - mnew);
      const float p0 = __expf(s0v - mnew);
      const float p1 = __expf(s1v - mnew);
      float psum = p0 + p1;
#pragma unroll
      for (int off = 1; off < 16; off <<= 1) psum += __shfl_xor(psum, off);
      lrow[r] = lrow[r] * corr + psum;
      mrow[r] = mnew;
#pragma unroll
      for (int vt = 0; vt < 8; ++vt) Oacc[vt][r] *= corr;
      Ps[wave * 640 + (r + hi8) * 40 + nc]      = (_Float16)p0;
      Ps[wave * 640 + (r + hi8) * 40 + 16 + nc] = (_Float16)p1;
    }

    // per-wave LDS round trip: ensure ds stores landed before fragment reads
    asm volatile("s_wait_dscnt 0" ::: "memory");

    // O += P * V : 8 n-tiles across dh=128
    v16h pa = lds_load_fragA(Ps + wave * 640, 40);
#pragma unroll
    for (int vt = 0; vt < 8; ++vt) {
      v16h b = lds_load_fragB(Vs + vt * 16, 136);
      Oacc[vt] = wmma_f16(pa, b, Oacc[vt]);
    }
  }

#pragma unroll
  for (int r = 0; r < 8; ++r) {
    const float inv = 1.0f / lrow[r];
    const int q = s0 + q0w + r + hi8;
#pragma unroll
    for (int vt = 0; vt < 8; ++vt)
      Oh[(size_t)q * DIM_ + h * DH_ + vt * 16 + nc] =
          (_Float16)(Oacc[vt][r] * inv);
  }
}

// ---------------------------------------------------------------------------
extern "C" void kernel_launch(void* const* d_in, const int* in_sizes, int n_in,
                              void* d_out, int out_size, void* d_ws, size_t ws_size,
                              hipStream_t stream) {
  (void)in_sizes; (void)n_in; (void)out_size; (void)ws_size;
  const float* X  = (const float*)d_in[0];
  const float* fc = (const float*)d_in[1];
  const float* fs = (const float*)d_in[2];
  const float* Wq = (const float*)d_in[3];
  const float* bq = (const float*)d_in[4];
  const float* Wk = (const float*)d_in[5];
  const float* bk = (const float*)d_in[6];
  const float* Wv = (const float*)d_in[7];
  const float* bv = (const float*)d_in[8];
  const float* Wo = (const float*)d_in[9];
  const float* bo = (const float*)d_in[10];
  const float* gq = (const float*)d_in[11];
  const float* gk = (const float*)d_in[12];

  char* ws = (char*)d_ws;
  const size_t f32mat = (size_t)S_ * DIM_ * sizeof(float);     // 32 MiB
  const size_t f16mat = (size_t)S_ * DIM_ * sizeof(_Float16);  // 16 MiB
  float*    Qf = (float*)ws;
  float*    Kf = (float*)(ws + f32mat);
  float*    Vf = (float*)(ws + 2 * f32mat);
  _Float16* Qh = (_Float16*)(ws + 3 * f32mat);
  _Float16* Kh = (_Float16*)(ws + 3 * f32mat + f16mat);
  _Float16* Vh = (_Float16*)(ws + 3 * f32mat + 2 * f16mat);
  _Float16* Oh = (_Float16*)(ws + 3 * f32mat + 3 * f16mat);

  dim3 gg(DIM_ / 128, S_ / 128);
  gemm_bias_kernel<false><<<gg, 256, 0, stream>>>((const void*)X, Wq, bq, Qf, S_, DIM_, DIM_);
  gemm_bias_kernel<false><<<gg, 256, 0, stream>>>((const void*)X, Wk, bk, Kf, S_, DIM_, DIM_);
  gemm_bias_kernel<false><<<gg, 256, 0, stream>>>((const void*)X, Wv, bv, Vf, S_, DIM_, DIM_);
  rmsnorm_rope_kernel<<<S_, 256, 0, stream>>>(Qf, gq, fc, fs, Qh);
  rmsnorm_rope_kernel<<<S_, 256, 0, stream>>>(Kf, gk, fc, fs, Kh);
  f32_to_f16_kernel<<<2048, 256, 0, stream>>>(Vf, Vh, S_ * DIM_);
  flash_attn_kernel<<<dim3(S_ / 64, H_), 128, 0, stream>>>(Qh, Kh, Vh, Oh);
  gemm_bias_kernel<true><<<gg, 256, 0, stream>>>((const void*)Oh, Wo, bo, (float*)d_out, S_, DIM_, DIM_);
}
